// SelfAttention1D_90168543412769
// MI455X (gfx1250) — matrix operand order repaired and checked by reading, not compiled, not run on hardware
//
#include <hip/hip_runtime.h>
#include <hip/hip_bf16.h>

#define B_   16
#define C_   256
#define CQK_ 32
#define L_   2048
#define LOG2E 1.4426950408889634f

typedef _Float16 v16h __attribute__((ext_vector_type(16)));
typedef _Float16 v8h  __attribute__((ext_vector_type(8)));
typedef float    v8f  __attribute__((ext_vector_type(8)));

union Frag16 { v16h v; v8h h[2]; _Float16 e[16]; };
union Half8  { v8h v; _Float16 e[8]; };

__device__ __forceinline__ float lanebcast(float x, int srcLane) {
  return __int_as_float(__builtin_amdgcn_ds_bpermute(srcLane << 2, __float_as_int(x)));
}

__device__ __forceinline__ v8f wmma_f16(const Frag16& a, const Frag16& b, v8f c) {
  return __builtin_amdgcn_wmma_f32_16x16x32_f16(false, a.v, false, b.v, (short)0, c,
                                                false, false);
}

// ---------------------------------------------------------------------------
// Kernel 0a: convert weights to f16, fragment-friendly row-major (o, 256).
// Wqh is pre-scaled by log2e so softmax can use native v_exp_f32 (exp2).
// ---------------------------------------------------------------------------
__global__ __launch_bounds__(256) void conv_w(
    const float* __restrict__ Wq, const float* __restrict__ Wk,
    const float* __restrict__ Wv, _Float16* __restrict__ Wqh,
    _Float16* __restrict__ Wkh, _Float16* __restrict__ Wvh) {
  int idx = blockIdx.x * 256 + threadIdx.x;       // 320*256 = 81920 = (32+32+256)*256
  if (idx < 8192)        Wqh[idx] = (_Float16)(Wq[idx] * LOG2E);
  else if (idx < 16384)  Wkh[idx - 8192] = (_Float16)Wk[idx - 8192];
  else                   Wvh[idx - 16384] = (_Float16)Wv[idx - 16384];
}

// ---------------------------------------------------------------------------
// Kernel 0b: transpose+convert x (B,C,L) f32 -> xh (B,L,256) f16 via LDS tiles,
// so x tiles load directly as WMMA B-operand fragments (contiguous in c).
// ---------------------------------------------------------------------------
__global__ __launch_bounds__(256) void conv_x(
    const float* __restrict__ x, _Float16* __restrict__ xh) {
  __shared__ float tile[64][65];
  const int tid = threadIdx.x;
  const int l0 = blockIdx.x * 64, c0 = blockIdx.y * 64, b = blockIdx.z;

#pragma unroll
  for (int i = 0; i < 16; ++i) {                   // 64c x 64l loads, coalesced in l
    int idx = i * 256 + tid;
    int cl = idx >> 6, ll = idx & 63;
    tile[cl][ll] = x[((size_t)b * 256 + c0 + cl) * L_ + l0 + ll];
  }
  __syncthreads();

  unsigned* xw = (unsigned*)xh;
#pragma unroll
  for (int i = 0; i < 8; ++i) {                    // pack c-pairs, coalesced in c
    int idx = i * 256 + tid;
    int ll = idx >> 5, cp = idx & 31;
    union { _Float16 h[2]; unsigned u; } p;
    p.h[0] = (_Float16)tile[2 * cp][ll];
    p.h[1] = (_Float16)tile[2 * cp + 1][ll];
    xw[((size_t)b * L_ + l0 + ll) * 128 + (c0 >> 1) + cp] = p.u;
  }
}

// ---------------------------------------------------------------------------
// Kernel 1: Q/K projection via WMMA.  One wave per 64 l-positions.
// D[o][l] = W(32x256) x xh^T.  Accumulator layout (lane=l, regs=o) stores
// straight into the (B,L,32) f16 row-major layout flash fragments expect.
// ---------------------------------------------------------------------------
__global__ __launch_bounds__(128, 1) void proj_qk(
    const _Float16* __restrict__ xh, const _Float16* __restrict__ Wqh,
    const _Float16* __restrict__ Wkh, const float* __restrict__ bq,
    const float* __restrict__ bk, _Float16* __restrict__ Qh,
    _Float16* __restrict__ Kh) {
  const int lane = threadIdx.x & 31, lm = lane & 15, hi = lane >> 4;
  const int wave = blockIdx.x * 4 + (threadIdx.x >> 5);   // 512 waves
  const int b = wave >> 5;                                // 32 l-tiles / batch
  const int l0 = (wave & 31) << 6;

  const v8f zero8 = {0.f, 0.f, 0.f, 0.f, 0.f, 0.f, 0.f, 0.f};
  v8f acc[4][4];                                          // [A-tile][N-tile]
#pragma unroll
  for (int a = 0; a < 4; ++a)
#pragma unroll
    for (int n = 0; n < 4; ++n) acc[a][n] = zero8;

  for (int kc = 0; kc < 8; ++kc) {
    const int cofs = kc * 32 + hi * 8;
    Frag16 aW[4];                                         // Q rows 0-15,16-31; K same
    aW[0].h[0] = *(const v8h*)(Wqh + lm * 256 + cofs);
    aW[0].h[1] = *(const v8h*)(Wqh + lm * 256 + cofs + 16);
    aW[1].h[0] = *(const v8h*)(Wqh + (16 + lm) * 256 + cofs);
    aW[1].h[1] = *(const v8h*)(Wqh + (16 + lm) * 256 + cofs + 16);
    aW[2].h[0] = *(const v8h*)(Wkh + lm * 256 + cofs);
    aW[2].h[1] = *(const v8h*)(Wkh + lm * 256 + cofs + 16);
    aW[3].h[0] = *(const v8h*)(Wkh + (16 + lm) * 256 + cofs);
    aW[3].h[1] = *(const v8h*)(Wkh + (16 + lm) * 256 + cofs + 16);
#pragma unroll
    for (int n = 0; n < 4; ++n) {
      Frag16 xB;
      xB.h[0] = *(const v8h*)(xh + ((size_t)b * L_ + l0 + n * 16 + lm) * 256 +
                              kc * 32 + hi * 8);
      xB.h[1] = *(const v8h*)(xh + ((size_t)b * L_ + l0 + n * 16 + lm) * 256 +
                              kc * 32 + hi * 8 + 16);
#pragma unroll
      for (int a = 0; a < 4; ++a) acc[a][n] = wmma_f16(aW[a], xB, acc[a][n]);
    }
  }

  float bias[4][8];                                       // o = {0,16}+8*hi+d
#pragma unroll
  for (int d = 0; d < 8; ++d) {
    bias[0][d] = bq[8 * hi + d] * LOG2E;
    bias[1][d] = bq[16 + 8 * hi + d] * LOG2E;
    bias[2][d] = bk[8 * hi + d];
    bias[3][d] = bk[16 + 8 * hi + d];
  }

#pragma unroll
  for (int n = 0; n < 4; ++n) {
    _Float16* qrow = Qh + ((size_t)b * L_ + l0 + n * 16 + lm) * 32;
    _Float16* krow = Kh + ((size_t)b * L_ + l0 + n * 16 + lm) * 32;
    Half8 r[4];
#pragma unroll
    for (int a = 0; a < 4; ++a)
#pragma unroll
      for (int d = 0; d < 8; ++d) r[a].e[d] = (_Float16)(acc[a][n][d] + bias[a][d]);
    *(v8h*)(qrow + 8 * hi)      = r[0].v;
    *(v8h*)(qrow + 16 + 8 * hi) = r[1].v;
    *(v8h*)(krow + 8 * hi)      = r[2].v;
    *(v8h*)(krow + 16 + 8 * hi) = r[3].v;
  }
}

// ---------------------------------------------------------------------------
// Kernel 2: V projection via WMMA.  One wave per 16 l-positions, 16 o-tiles.
// Output Vh (B, 256, L) channel-major for the flash V B-fragments.
// ---------------------------------------------------------------------------
__global__ __launch_bounds__(128, 1) void proj_v(
    const _Float16* __restrict__ xh, const _Float16* __restrict__ Wvh,
    const float* __restrict__ bv, _Float16* __restrict__ Vh) {
  const int lane = threadIdx.x & 31, lm = lane & 15, hi = lane >> 4;
  const int wave = blockIdx.x * 4 + (threadIdx.x >> 5);   // 2048 waves
  const int b = wave >> 7;
  const int l0 = (wave & 127) << 4;

  const v8f zero8 = {0.f, 0.f, 0.f, 0.f, 0.f, 0.f, 0.f, 0.f};
  v8f acc[16];
#pragma unroll
  for (int t = 0; t < 16; ++t) acc[t] = zero8;

  for (int kc = 0; kc < 8; ++kc) {
    Frag16 xB;
    xB.h[0] = *(const v8h*)(xh + ((size_t)b * L_ + l0 + lm) * 256 + kc * 32 + hi * 8);
    xB.h[1] = *(const v8h*)(xh + ((size_t)b * L_ + l0 + lm) * 256 + kc * 32 + hi * 8 + 16);
    const int cofs = kc * 32 + hi * 8;
#pragma unroll
    for (int t = 0; t < 16; ++t) {
      Frag16 a;
      a.h[0] = *(const v8h*)(Wvh + (t * 16 + lm) * 256 + cofs);
      a.h[1] = *(const v8h*)(Wvh + (t * 16 + lm) * 256 + cofs + 16);
      acc[t] = wmma_f16(a, xB, acc[t]);
    }
  }

#pragma unroll
  for (int t = 0; t < 16; ++t) {
#pragma unroll
    for (int d = 0; d < 8; ++d) {
      int o = t * 16 + 8 * hi + d;
      Vh[((size_t)b * 256 + o) * L_ + l0 + lm] = (_Float16)(acc[t][d] + bv[o]);
    }
  }
}

// ---------------------------------------------------------------------------
// Kernel 3: fused flash attention + residual (unchanged from round 1).
// One wave32 per 16-query tile; S^T = K_tile x Q^T so the S accumulator layout
// equals the PV WMMA's A-operand layout (P needs no cross-lane movement).
// ---------------------------------------------------------------------------
__global__ __launch_bounds__(128, 1) void flash_attn(
    const _Float16* __restrict__ Qh, const _Float16* __restrict__ Kh,
    const _Float16* __restrict__ Vh, const float* __restrict__ x,
    const float* __restrict__ gamma, float* __restrict__ out) {
  const int lane = threadIdx.x & 31;
  const int lm   = lane & 15;
  const int hi   = lane >> 4;
  const int wave = blockIdx.x * 4 + (threadIdx.x >> 5);
  const int b    = wave >> 7;
  const int qt   = (wave & 127) << 4;

  const v8f zero8 = {0.f, 0.f, 0.f, 0.f, 0.f, 0.f, 0.f, 0.f};

  Frag16 qB;
  qB.h[0] = *(const v8h*)(Qh + ((size_t)(b * L_ + qt + lm)) * 32 + hi * 8);
  qB.h[1] = *(const v8h*)(Qh + ((size_t)(b * L_ + qt + lm)) * 32 + hi * 8 + 16);

  v8f O[16];
#pragma unroll
  for (int t = 0; t < 16; ++t) O[t] = zero8;

  float mrun = -3.0e38f, lrun = 0.f;

  const size_t kbatch = (size_t)b * L_ * 32;
  const size_t vbatch = (size_t)b * 256 * L_;

  for (int jt = 0; jt < L_; jt += 32) {
    Frag16 kA0, kA1;
    const _Float16* kp0 = Kh + kbatch + (size_t)(jt + lm) * 32 + hi * 8;
    const _Float16* kp1 = kp0 + 16 * 32;
    kA0.h[0] = *(const v8h*)(kp0);
    kA0.h[1] = *(const v8h*)(kp0 + 16);
    kA1.h[0] = *(const v8h*)(kp1);
    kA1.h[1] = *(const v8h*)(kp1 + 16);

    v8f s0 = wmma_f16(kA0, qB, zero8);
    v8f s1 = wmma_f16(kA1, qB, zero8);

    float mloc = -3.0e38f;
#pragma unroll
    for (int d = 0; d < 8; ++d) {
      mloc = fmaxf(mloc, s0[d]);
      mloc = fmaxf(mloc, s1[d]);
    }
    mloc = fmaxf(mloc, lanebcast(mloc, lane ^ 16));
    float mnew  = fmaxf(mrun, mloc);
    float alpha = __builtin_amdgcn_exp2f(mrun - mnew);

    Frag16 pA;
    float psum = 0.f;
#pragma unroll
    for (int d = 0; d < 8; ++d) {
      float p0 = __builtin_amdgcn_exp2f(s0[d] - mnew);
      float p1 = __builtin_amdgcn_exp2f(s1[d] - mnew);
      psum += p0 + p1;
      pA.e[d]     = (_Float16)p0;
      pA.e[8 + d] = (_Float16)p1;
    }
    psum += lanebcast(psum, lane ^ 16);
    lrun = lrun * alpha + psum;
    mrun = mnew;

    float asel[8];
#pragma unroll
    for (int d = 0; d < 8; ++d) asel[d] = lanebcast(alpha, d + (hi << 3));

#pragma unroll
    for (int t = 0; t < 16; ++t) {
      Frag16 vB;
      vB.h[0] = *(const v8h*)(Vh + vbatch + (size_t)(t * 16 + lm) * L_ + jt + hi * 8);
      vB.h[1] = *(const v8h*)(Vh + vbatch + (size_t)(t * 16 + lm) * L_ + jt + hi * 8 + 16);
      v8f o = O[t];
#pragma unroll
      for (int d = 0; d < 8; ++d) o[d] *= asel[d];
      O[t] = wmma_f16(pA, vB, o);
    }
  }

  float rl = 1.0f / lrun;
  float rsel[8];
#pragma unroll
  for (int d = 0; d < 8; ++d) rsel[d] = lanebcast(rl, d + (hi << 3));
  const float g = gamma[0];

#pragma unroll
  for (int t = 0; t < 16; ++t) {
#pragma unroll
    for (int d = 0; d < 8; ++d) {
      size_t idx = ((size_t)b * 256 + t * 16 + lm) * L_ + (qt + d + (hi << 3));
      out[idx] = fmaf(g, O[t][d] * rsel[d], x[idx]);
    }
  }
}

// ---------------------------------------------------------------------------
extern "C" void kernel_launch(void* const* d_in, const int* in_sizes, int n_in,
                              void* d_out, int out_size, void* d_ws, size_t ws_size,
                              hipStream_t stream) {
  (void)in_sizes; (void)n_in; (void)out_size; (void)ws_size;
  const float* x     = (const float*)d_in[0];
  const float* Wq    = (const float*)d_in[1];
  const float* bq    = (const float*)d_in[2];
  const float* Wk    = (const float*)d_in[3];
  const float* bk    = (const float*)d_in[4];
  const float* Wv    = (const float*)d_in[5];
  const float* bv    = (const float*)d_in[6];
  const float* gamma = (const float*)d_in[7];
  float* out = (float*)d_out;

  _Float16* xh  = (_Float16*)d_ws;                        // B*L*256   (16 MB)
  _Float16* Vh  = xh  + (size_t)B_ * L_ * 256;            // B*256*L   (16 MB)
  _Float16* Qh  = Vh  + (size_t)B_ * 256 * L_;            // B*L*32    ( 2 MB)
  _Float16* Kh  = Qh  + (size_t)B_ * L_ * 32;             // B*L*32    ( 2 MB)
  _Float16* Wqh = Kh  + (size_t)B_ * L_ * 32;             // 32*256
  _Float16* Wkh = Wqh + 32 * 256;                         // 32*256
  _Float16* Wvh = Wkh + 32 * 256;                         // 256*256

  conv_w<<<320, 256, 0, stream>>>(Wq, Wk, Wv, Wqh, Wkh, Wvh);
  conv_x<<<dim3(32, 4, 16), 256, 0, stream>>>(x, xh);
  proj_qk<<<128, 128, 0, stream>>>(xh, Wqh, Wkh, bq, bk, Qh, Kh);
  proj_v<<<512, 128, 0, stream>>>(xh, Wvh, bv, Vh);
  flash_attn<<<512, 128, 0, stream>>>(Qh, Kh, Vh, x, gamma, out);
}
